// GenLSTMp_9938554323548
// MI455X (gfx1250) — matrix operand order repaired and checked by reading, not compiled, not run on hardware
//
#include <hip/hip_runtime.h>

typedef __attribute__((ext_vector_type(16))) _Float16 v16h;
typedef __attribute__((ext_vector_type(8)))  float    v8f;

// ---------------------------------------------------------------------------
// Problem constants: B=4096, T=256, ND=16, SD=8, H=64, IN=24, K_concat=88->96
// ---------------------------------------------------------------------------

// Workspace layout in _Float16 units (packed B-fragments):
//   WXH : K=96  N=256 -> 3*16*512 = 24576 halves
//   WM1 : K=64  N=64  -> 2*4*512  = 4096
//   WM2 : 4096, WM3 (N pad 8->16): 2*1*512 = 1024, then WV1/WV2/WV3 likewise.
#define OFF_WXH 0
#define OFF_WM1 24576
#define OFF_WM2 28672
#define OFF_WM3 32768
#define OFF_WV1 33792
#define OFF_WV2 37888
#define OFF_WV3 41984
#define OFF_WCOMB_BYTES 131072   // float staging of concat(Wx,Wh,0) [96][256]

static __device__ __forceinline__ v8f wmma_f16(v16h a, v16h b, v8f c) {
  return __builtin_amdgcn_wmma_f32_16x16x32_f16(false, a, false, b, (short)0, c,
                                                false, false);
}

// A-fragment (16x32 f16, M=16 rows of a row-major f16 LDS buffer):
// lanes 0-15 (row m=lane):  halves 0..7 = K k0+0..7,  halves 8..15 = K k0+16..23
// lanes 16-31 (row m=l-16): halves 0..7 = K k0+8..15, halves 8..15 = K k0+24..31
static __device__ __forceinline__ v16h load_afrag(const _Float16* buf, int stride,
                                                  int k0, int lane) {
  const int m  = lane & 15;
  const int hi = lane >> 4;
  const _Float16* p = buf + m * stride + k0 + hi * 8;
  v16h a;
#pragma unroll
  for (int i = 0; i < 8; ++i) { a[i] = p[i]; a[8 + i] = p[16 + i]; }
  return a;
}

// B-fragment: pre-packed so each lane's 16 halves are contiguous (32B aligned).
static __device__ __forceinline__ v16h load_bfrag(const _Float16* pack, int frag,
                                                  int lane) {
  return *(const v16h*)(pack + ((frag << 5) + lane) * 16);
}

// Hardware reciprocal (v_rcp_f32) instead of IEEE divide sequences.
static __device__ __forceinline__ float fast_rcp(float x) {
  return __builtin_amdgcn_rcpf(x);
}
static __device__ __forceinline__ float sigm(float x) {
  return fast_rcp(1.0f + __expf(-x));
}
static __device__ __forceinline__ float tanh_fast(float x) {
  return 1.0f - 2.0f * fast_rcp(__expf(2.0f * x) + 1.0f);
}

static __device__ __forceinline__ void store_relu_tile(_Float16* buf, int stride,
                                                       int colBase, v8f acc,
                                                       int lane) {
  const int n  = lane & 15;
  const int hi = lane >> 4;
#pragma unroll
  for (int r = 0; r < 8; ++r) {
    float v = acc[r];
    v = v > 0.0f ? v : 0.0f;
    buf[(hi * 8 + r) * stride + colBase + n] = (_Float16)v;
  }
}

// ---------------------------------------------------------------------------
// Prep kernels: stage concat(Wx,Wh,0) and pack all weights into B-fragment
// layout: packed[((kb*NB+nb)*32 + lane)*16 + j] = W[kb*32 + (j<8? 8*(lane>>4)+j : 16+8*(lane>>4)+j-8)]
//                                                  [nb*16 + (lane&15)]
// ---------------------------------------------------------------------------
__global__ void build_wcomb(const float* __restrict__ Wx,
                            const float* __restrict__ Wh,
                            float* __restrict__ Wcomb) {
  int i = blockIdx.x * 256 + threadIdx.x;   // 96*256
  if (i >= 96 * 256) return;
  int row = i >> 8, col = i & 255;
  float v = 0.0f;
  if (row < 24)      v = Wx[row * 256 + col];
  else if (row < 88) v = Wh[(row - 24) * 256 + col];
  Wcomb[i] = v;
}

__global__ void pack_weight(const float* __restrict__ src, int Ksrc, int Nsrc,
                            int KB, int NB, _Float16* __restrict__ dst) {
  int p = blockIdx.x * 256 + threadIdx.x;
  int total = KB * NB * 512;
  if (p >= total) return;
  int j    = p & 15;
  int lane = (p >> 4) & 31;
  int nb   = (p >> 9) % NB;
  int kb   = p / (NB * 512);
  int k = kb * 32 + ((j < 8) ? (lane >> 4) * 8 + j : 16 + (lane >> 4) * 8 + (j - 8));
  int n = nb * 16 + (lane & 15);
  float v = (k < Ksrc && n < Nsrc) ? src[k * Nsrc + n] : 0.0f;
  dst[p] = (_Float16)v;
}

// ---------------------------------------------------------------------------
// Main fused kernel: each wave32 owns 16 batch rows for all 255 steps.
// ---------------------------------------------------------------------------
__global__ __launch_bounds__(256) void genlstm_main(
    const float* __restrict__ noise, const float* __restrict__ eps,
    const float* __restrict__ bz,
    const float* __restrict__ bm1, const float* __restrict__ bm2,
    const float* __restrict__ bm3,
    const float* __restrict__ bv1, const float* __restrict__ bv2,
    const float* __restrict__ bv3,
    const _Float16* __restrict__ packs,
    float* __restrict__ out) {
  // Per-wave LDS: inb[16][96] = [x(8) | noise(16) | h(64) | pad(8)], act[16][64]
  __shared__ _Float16 s_inb[8][16 * 96];
  __shared__ _Float16 s_act[8][16 * 64];

  const int lane  = threadIdx.x & 31;
  const int wave  = threadIdx.x >> 5;
  const int hi    = lane >> 4;
  const int n     = lane & 15;
  const int bbase = (blockIdx.x * 8 + wave) * 16;

  _Float16* inb = s_inb[wave];
  _Float16* act = s_act[wave];

  for (int i = lane; i < 16 * 96; i += 32) inb[i] = (_Float16)0.0f;

  // Bias preload (per output column n of each 16-wide tile).
  float bzr[16];
#pragma unroll
  for (int q = 0; q < 16; ++q) bzr[q] = bz[q * 16 + n];
  float b1m[4], b2m[4], b1v[4], b2v[4];
#pragma unroll
  for (int q = 0; q < 4; ++q) {
    b1m[q] = bm1[q * 16 + n]; b2m[q] = bm2[q * 16 + n];
    b1v[q] = bv1[q * 16 + n]; b2v[q] = bv2[q * 16 + n];
  }
  const float b3m = (n < 8) ? bm3[n] : 0.0f;
  const float b3v = (n < 8) ? bv3[n] : 0.0f;

  const _Float16* Wxh  = packs + OFF_WXH;
  const _Float16* Wm1p = packs + OFF_WM1;
  const _Float16* Wm2p = packs + OFF_WM2;
  const _Float16* Wm3p = packs + OFF_WM3;
  const _Float16* Wv1p = packs + OFF_WV1;
  const _Float16* Wv2p = packs + OFF_WV2;
  const _Float16* Wv3p = packs + OFF_WV3;

  // Cell state kept in WMMA C layout: c4[q][r] is (m = hi*8+r, j = q*16+n).
  v8f c4[4];
#pragma unroll
  for (int q = 0; q < 4; ++q)
#pragma unroll
    for (int r = 0; r < 8; ++r) c4[q][r] = 0.0f;

  float cum[8];
#pragma unroll
  for (int r = 0; r < 8; ++r) cum[r] = 0.0f;

  // Output row t=0 is x0 = 0 (cumsum of zeros).
  if (n < 8) {
#pragma unroll
    for (int r = 0; r < 8; ++r) {
      int m = hi * 8 + r;
      out[(size_t)(bbase + m) * 256 * 8 + n] = 0.0f;
    }
  }

  for (int t = 0; t < 255; ++t) {
    // Defeat LICM of the (loop-invariant) packed-weight B-fragment loads:
    // keeping them in-loop avoids ~670 VGPRs of hoisted live range that the
    // compiler otherwise spills to scratch. The packs are L2-resident.
    asm volatile("" ::: "memory");

    // Stage this step's noise into inb cols 8..23 (f32 -> f16).
    {
      const float* np = noise + ((size_t)(bbase + n) * 255 + t) * 16 + hi * 8;
      _Float16* dp = inb + n * 96 + 8 + hi * 8;
#pragma unroll
      for (int i = 0; i < 8; ++i) dp[i] = (_Float16)np[i];
    }

    // A operands for the fused gate GEMM: [x|noise|h|pad] K=96, 3 chunks.
    v16h aG[3];
#pragma unroll
    for (int kb = 0; kb < 3; ++kb) aG[kb] = load_afrag(inb, 96, kb * 32, lane);

    v8f si[4];
    // i gates (z cols 0..63)
#pragma unroll
    for (int q = 0; q < 4; ++q) {
      v8f acc;
#pragma unroll
      for (int r = 0; r < 8; ++r) acc[r] = bzr[q];
#pragma unroll
      for (int kb = 0; kb < 3; ++kb)
        acc = wmma_f16(aG[kb], load_bfrag(Wxh, kb * 16 + q, lane), acc);
#pragma unroll
      for (int r = 0; r < 8; ++r) si[q][r] = sigm(acc[r]);
    }
    // g gates (z cols 128..191): si *= tanh(g)
#pragma unroll
    for (int q = 0; q < 4; ++q) {
      v8f acc;
#pragma unroll
      for (int r = 0; r < 8; ++r) acc[r] = bzr[8 + q];
#pragma unroll
      for (int kb = 0; kb < 3; ++kb)
        acc = wmma_f16(aG[kb], load_bfrag(Wxh, kb * 16 + 8 + q, lane), acc);
#pragma unroll
      for (int r = 0; r < 8; ++r) si[q][r] *= tanh_fast(acc[r]);
    }
    // f gates (z cols 64..127): c = sig(f)*c + sig(i)*tanh(g)
#pragma unroll
    for (int q = 0; q < 4; ++q) {
      v8f acc;
#pragma unroll
      for (int r = 0; r < 8; ++r) acc[r] = bzr[4 + q];
#pragma unroll
      for (int kb = 0; kb < 3; ++kb)
        acc = wmma_f16(aG[kb], load_bfrag(Wxh, kb * 16 + 4 + q, lane), acc);
#pragma unroll
      for (int r = 0; r < 8; ++r) c4[q][r] = sigm(acc[r]) * c4[q][r] + si[q][r];
    }
    // o gates (z cols 192..255): h = sig(o)*tanh(c), write h into inb cols 24..87
#pragma unroll
    for (int q = 0; q < 4; ++q) {
      v8f acc;
#pragma unroll
      for (int r = 0; r < 8; ++r) acc[r] = bzr[12 + q];
#pragma unroll
      for (int kb = 0; kb < 3; ++kb)
        acc = wmma_f16(aG[kb], load_bfrag(Wxh, kb * 16 + 12 + q, lane), acc);
#pragma unroll
      for (int r = 0; r < 8; ++r) {
        float hval = sigm(acc[r]) * tanh_fast(c4[q][r]);
        inb[(hi * 8 + r) * 96 + 24 + q * 16 + n] = (_Float16)hval;
      }
    }

    v8f mu, lv;
    // ---- mu MLP: relu(h@Wm1+b1) -> relu(@Wm2+b2) -> @Wm3+b3 ----
    {
      v16h a0 = load_afrag(inb, 96, 24, lane);
      v16h a1 = load_afrag(inb, 96, 56, lane);
      v8f l1[4];
#pragma unroll
      for (int q = 0; q < 4; ++q) {
        v8f acc;
#pragma unroll
        for (int r = 0; r < 8; ++r) acc[r] = b1m[q];
        acc = wmma_f16(a0, load_bfrag(Wm1p, q, lane), acc);
        acc = wmma_f16(a1, load_bfrag(Wm1p, 4 + q, lane), acc);
        l1[q] = acc;
      }
#pragma unroll
      for (int q = 0; q < 4; ++q) store_relu_tile(act, 64, q * 16, l1[q], lane);
      v16h c0 = load_afrag(act, 64, 0, lane);
      v16h c1 = load_afrag(act, 64, 32, lane);
      v8f l2[4];
#pragma unroll
      for (int q = 0; q < 4; ++q) {
        v8f acc;
#pragma unroll
        for (int r = 0; r < 8; ++r) acc[r] = b2m[q];
        acc = wmma_f16(c0, load_bfrag(Wm2p, q, lane), acc);
        acc = wmma_f16(c1, load_bfrag(Wm2p, 4 + q, lane), acc);
        l2[q] = acc;
      }
#pragma unroll
      for (int q = 0; q < 4; ++q) store_relu_tile(act, 64, q * 16, l2[q], lane);
      v16h d0 = load_afrag(act, 64, 0, lane);
      v16h d1 = load_afrag(act, 64, 32, lane);
      v8f acc;
#pragma unroll
      for (int r = 0; r < 8; ++r) acc[r] = b3m;
      acc = wmma_f16(d0, load_bfrag(Wm3p, 0, lane), acc);
      acc = wmma_f16(d1, load_bfrag(Wm3p, 1, lane), acc);
      mu = acc;
    }
    // ---- logvar MLP (same structure, act buffer reused) ----
    {
      v16h a0 = load_afrag(inb, 96, 24, lane);
      v16h a1 = load_afrag(inb, 96, 56, lane);
      v8f l1[4];
#pragma unroll
      for (int q = 0; q < 4; ++q) {
        v8f acc;
#pragma unroll
        for (int r = 0; r < 8; ++r) acc[r] = b1v[q];
        acc = wmma_f16(a0, load_bfrag(Wv1p, q, lane), acc);
        acc = wmma_f16(a1, load_bfrag(Wv1p, 4 + q, lane), acc);
        l1[q] = acc;
      }
#pragma unroll
      for (int q = 0; q < 4; ++q) store_relu_tile(act, 64, q * 16, l1[q], lane);
      v16h c0 = load_afrag(act, 64, 0, lane);
      v16h c1 = load_afrag(act, 64, 32, lane);
      v8f l2[4];
#pragma unroll
      for (int q = 0; q < 4; ++q) {
        v8f acc;
#pragma unroll
        for (int r = 0; r < 8; ++r) acc[r] = b2v[q];
        acc = wmma_f16(c0, load_bfrag(Wv2p, q, lane), acc);
        acc = wmma_f16(c1, load_bfrag(Wv2p, 4 + q, lane), acc);
        l2[q] = acc;
      }
#pragma unroll
      for (int q = 0; q < 4; ++q) store_relu_tile(act, 64, q * 16, l2[q], lane);
      v16h d0 = load_afrag(act, 64, 0, lane);
      v16h d1 = load_afrag(act, 64, 32, lane);
      v8f acc;
#pragma unroll
      for (int r = 0; r < 8; ++r) acc[r] = b3v;
      acc = wmma_f16(d0, load_bfrag(Wv3p, 0, lane), acc);
      acc = wmma_f16(d1, load_bfrag(Wv3p, 1, lane), acc);
      lv = acc;
    }

    // Sample, cumsum, emit, and feed x back for the next step.
    if (n < 8) {
#pragma unroll
      for (int r = 0; r < 8; ++r) {
        int m = hi * 8 + r;
        float e = eps[((size_t)(bbase + m) * 255 + t) * 8 + n];
        float x = mu[r] + __expf(0.5f * lv[r]) * e;
        cum[r] += x;
        out[((size_t)(bbase + m) * 256 + (t + 1)) * 8 + n] = cum[r];
        inb[m * 96 + n] = (_Float16)x;
      }
    }
  }
}

extern "C" void kernel_launch(void* const* d_in, const int* in_sizes, int n_in,
                              void* d_out, int out_size, void* d_ws,
                              size_t ws_size, hipStream_t stream) {
  const float* noise = (const float*)d_in[0];
  const float* eps   = (const float*)d_in[1];
  const float* Wx    = (const float*)d_in[2];
  const float* Wh    = (const float*)d_in[3];
  const float* b     = (const float*)d_in[4];
  const float* Wm1   = (const float*)d_in[5];
  const float* bm1   = (const float*)d_in[6];
  const float* Wm2   = (const float*)d_in[7];
  const float* bm2   = (const float*)d_in[8];
  const float* Wm3   = (const float*)d_in[9];
  const float* bm3   = (const float*)d_in[10];
  const float* Wv1   = (const float*)d_in[11];
  const float* bv1   = (const float*)d_in[12];
  const float* Wv2   = (const float*)d_in[13];
  const float* bv2   = (const float*)d_in[14];
  const float* Wv3   = (const float*)d_in[15];
  const float* bv3   = (const float*)d_in[16];

  _Float16* packs = (_Float16*)d_ws;
  float* Wcomb = (float*)((char*)d_ws + OFF_WCOMB_BYTES);

  build_wcomb<<<96, 256, 0, stream>>>(Wx, Wh, Wcomb);
  pack_weight<<<96, 256, 0, stream>>>(Wcomb, 96, 256, 3, 16, packs + OFF_WXH);
  pack_weight<<<16, 256, 0, stream>>>(Wm1, 64, 64, 2, 4, packs + OFF_WM1);
  pack_weight<<<16, 256, 0, stream>>>(Wm2, 64, 64, 2, 4, packs + OFF_WM2);
  pack_weight<<<4, 256, 0, stream>>>(Wm3, 64, 8, 2, 1, packs + OFF_WM3);
  pack_weight<<<16, 256, 0, stream>>>(Wv1, 64, 64, 2, 4, packs + OFF_WV1);
  pack_weight<<<16, 256, 0, stream>>>(Wv2, 64, 64, 2, 4, packs + OFF_WV2);
  pack_weight<<<4, 256, 0, stream>>>(Wv3, 64, 8, 2, 1, packs + OFF_WV3);

  // 32 blocks * 8 waves * 16 batch rows = 4096 sequences.
  genlstm_main<<<32, 256, 0, stream>>>(noise, eps, b, bm1, bm2, bm3, bv1, bv2,
                                       bv3, packs, (float*)d_out);
}
